// AttentionLSTM_24910810317052
// MI455X (gfx1250) — matrix operand order repaired and checked by reading, not compiled, not run on hardware
//
#include <hip/hip_runtime.h>

// Problem sizes (fixed by reference): N=256, T=128, D=1024, H=1024, 4H=4096
#define BN 256
#define BT 128
#define BD 1024
#define BH 1024
#define FH 4096
#define KCAT 2048

typedef __attribute__((ext_vector_type(16))) __bf16 v16bf;
typedef __attribute__((ext_vector_type(8)))  float  v8f;
typedef __attribute__((ext_vector_type(4)))  unsigned int u32x4;

union Frag16 { v16bf v; u32x4 q[2]; };
static_assert(sizeof(Frag16) == 32, "frag size");

__device__ __forceinline__ unsigned short f2bf(float f) {
  unsigned int u = __float_as_uint(f);
  u += 0x7FFFu + ((u >> 16) & 1u);   // round-to-nearest-even
  return (unsigned short)(u >> 16);
}

// A fragment (16x32 bf16, M x K), A row-major with leading dim lda (elements).
// lane: m = lane&15, kh = lane>>4 ; elems0-7 = K[kh*8..], elems8-15 = K[16+kh*8..]
__device__ __forceinline__ v16bf load_a(const unsigned short* A, int lda,
                                        int row0, int k0, int lane) {
  int m = lane & 15, kh = lane >> 4;
  const unsigned short* p = A + (size_t)(row0 + m) * lda + k0 + kh * 8;
  Frag16 f;
  f.q[0] = *reinterpret_cast<const u32x4*>(p);
  f.q[1] = *reinterpret_cast<const u32x4*>(p + 16);
  return f.v;
}

// B fragment (32x16 bf16, K x N) sourced from pre-transposed W^T (Ncols x K)
// row-major. lane: n = lane&15, kh = lane>>4 ; elems0-7 = K[kh*8..], elems8-15 = K[16+kh*8..]
__device__ __forceinline__ v16bf load_b(const unsigned short* WT, int ldk,
                                        int col0, int k0, int lane) {
  int n = lane & 15, kb = (lane >> 4) * 8;
  const unsigned short* p = WT + (size_t)(col0 + n) * ldk + k0 + kb;
  Frag16 f;
  f.q[0] = *reinterpret_cast<const u32x4*>(p);
  f.q[1] = *reinterpret_cast<const u32x4*>(p + 16);
  return f.v;
}

// ---------- one-time prep kernels ----------

// out[j*outStride + kOff + k] = bf16(W[k*4096 + j]);  j in [0,4096), k in [0,Kdim)
__global__ __launch_bounds__(256) void k_transpose_cvt(
    const float* __restrict__ W, unsigned short* __restrict__ out,
    int Kdim, int outStride, int kOff) {
  int idx = blockIdx.x * 256 + threadIdx.x;
  int total = FH * Kdim;
  if (idx >= total) return;
  int j = idx / Kdim;
  int k = idx - j * Kdim;
  out[(size_t)j * outStride + kOff + k] = f2bf(W[(size_t)k * FH + j]);
}

__global__ __launch_bounds__(256) void k_cvt_bf16(
    const float* __restrict__ in, unsigned short* __restrict__ out, size_t total) {
  size_t i = (size_t)blockIdx.x * 256 + threadIdx.x;
  if (i < total) out[i] = f2bf(in[i]);
}

// h0 = mean over 16 spatial; c0 = h0; also seed bf16 h-half of hcat0
__global__ __launch_bounds__(256) void k_init(
    const float* __restrict__ Af, float* __restrict__ h, float* __restrict__ c,
    unsigned short* __restrict__ hcat0) {
  int idx = blockIdx.x * 256 + threadIdx.x;         // over N*H
  const float* a = Af + (size_t)idx * 16;
  float s = 0.f;
  #pragma unroll
  for (int l = 0; l < 16; ++l) s += a[l];
  s *= (1.f / 16.f);
  h[idx] = s;
  c[idx] = s;
  int n = idx >> 10, k = idx & 1023;
  hcat0[(size_t)n * KCAT + k] = f2bf(s);
}

// ---------- big input projection: Xp = x_bf16 @ Wx + b  (M=32768,K=1024,N=4096)
// block = 256 threads = 8 waves; wave -> 32 rows (2 M-subtiles) x 64 cols (4 N-subtiles)
// => 8 accumulators/wave, 12 b128 loads per 8 WMMA.
__global__ __launch_bounds__(256) void k_xproj(
    const unsigned short* __restrict__ xbf, const unsigned short* __restrict__ WxT,
    const float* __restrict__ bias, float* __restrict__ Xp) {
  int lane = threadIdx.x & 31, wave = threadIdx.x >> 5;
  int row0 = blockIdx.x * 256 + wave * 32;   // [0, 32768)
  int col0 = blockIdx.y * 64;                // [0, 4096)
  v8f acc[2][4];
  #pragma unroll
  for (int mi = 0; mi < 2; ++mi)
    #pragma unroll
    for (int s = 0; s < 4; ++s) acc[mi][s] = (v8f){0,0,0,0,0,0,0,0};

  for (int k0 = 0; k0 < BD; k0 += 32) {
    v16bf a0 = load_a(xbf, BD, row0,      k0, lane);
    v16bf a1 = load_a(xbf, BD, row0 + 16, k0, lane);
    #pragma unroll
    for (int s = 0; s < 4; ++s) {
      v16bf b = load_b(WxT, BD, col0 + s * 16, k0, lane);
      acc[0][s] = __builtin_amdgcn_wmma_f32_16x16x32_bf16(false, a0, false, b, (short)0, acc[0][s], false, false);
      acc[1][s] = __builtin_amdgcn_wmma_f32_16x16x32_bf16(false, a1, false, b, (short)0, acc[1][s], false, false);
    }
  }
  int nn = lane & 15, hi = lane >> 4;
  #pragma unroll
  for (int mi = 0; mi < 2; ++mi) {
    #pragma unroll
    for (int s = 0; s < 4; ++s) {
      int col = col0 + s * 16 + nn;
      float bv = bias[col];
      #pragma unroll
      for (int r = 0; r < 8; ++r) {
        int m = mi * 16 + r + hi * 8;
        Xp[(size_t)(row0 + m) * FH + col] = acc[mi][s][r] + bv;
      }
    }
  }
}

// ---------- per-step attention (VALU): logits->softmax->attn, writes attn half of hcat_in
__global__ __launch_bounds__(256) void k_attn(
    const float* __restrict__ Af, const float* __restrict__ h,
    unsigned short* __restrict__ hcat_in) {
  __shared__ float red[16 * 256];
  __shared__ float wts[16];
  int n = blockIdx.x, tid = threadIdx.x;
  float part[16];
  #pragma unroll
  for (int l = 0; l < 16; ++l) part[l] = 0.f;
  #pragma unroll
  for (int j = 0; j < 4; ++j) {
    int k = tid * 4 + j;
    float hv = h[(size_t)n * BH + k];
    const float* a = Af + ((size_t)n * BH + k) * 16;
    #pragma unroll
    for (int l = 0; l < 16; ++l) part[l] += hv * a[l];
  }
  #pragma unroll
  for (int l = 0; l < 16; ++l) red[l * 256 + tid] = part[l];
  for (int s = 128; s > 0; s >>= 1) {
    __syncthreads();
    if (tid < s) {
      #pragma unroll
      for (int l = 0; l < 16; ++l) red[l * 256 + tid] += red[l * 256 + tid + s];
    }
  }
  __syncthreads();
  if (tid == 0) {
    const float scale = 0.03125f;  // 1/sqrt(1024)
    float v[16], mx = -1e30f;
    #pragma unroll
    for (int l = 0; l < 16; ++l) { v[l] = red[l * 256] * scale; mx = fmaxf(mx, v[l]); }
    float ssum = 0.f, e[16];
    #pragma unroll
    for (int l = 0; l < 16; ++l) { e[l] = __expf(v[l] - mx); ssum += e[l]; }
    float inv = 1.f / ssum;
    #pragma unroll
    for (int l = 0; l < 16; ++l) wts[l] = e[l] * inv;
  }
  __syncthreads();
  #pragma unroll
  for (int j = 0; j < 4; ++j) {
    int k = tid * 4 + j;
    const float* a = Af + ((size_t)n * BH + k) * 16;
    float s = 0.f;
    #pragma unroll
    for (int l = 0; l < 16; ++l) s += a[l] * wts[l];
    hcat_in[(size_t)n * KCAT + BH + k] = f2bf(s);
  }
}

// ---------- per-step recurrent GEMM + fused LSTM gates
// pre = Xp[:,t,:] + [h|attn] @ [Wh;Wattn]   (M=256, K=2048, cols grouped 4 gates x 16)
// 16 rows/wave keeps 128 blocks in flight -- the serial chain is occupancy-starved,
// and the 16MB of recurrent weights stay hot in the 192MB L2 across all 128 steps.
__global__ __launch_bounds__(256) void k_gate(
    const unsigned short* __restrict__ hcat_in, const unsigned short* __restrict__ WcatT,
    const float* __restrict__ Xp, float* __restrict__ h, float* __restrict__ c,
    unsigned short* __restrict__ hcat_out, float* __restrict__ out, int t) {
  int lane = threadIdx.x & 31, wave = threadIdx.x >> 5;
  int row0 = blockIdx.x * 128 + wave * 16;   // batch rows [0,256)
  int c0 = blockIdx.y * 16;                  // h-column tile [0,1024)
  v8f ai = {0,0,0,0,0,0,0,0}, af = ai, ao = ai, ag = ai;
  for (int k0 = 0; k0 < KCAT; k0 += 32) {
    v16bf a = load_a(hcat_in, KCAT, row0, k0, lane);
    v16bf bi = load_b(WcatT, KCAT, 0 * BH + c0, k0, lane);
    ai = __builtin_amdgcn_wmma_f32_16x16x32_bf16(false, a, false, bi, (short)0, ai, false, false);
    v16bf bf = load_b(WcatT, KCAT, 1 * BH + c0, k0, lane);
    af = __builtin_amdgcn_wmma_f32_16x16x32_bf16(false, a, false, bf, (short)0, af, false, false);
    v16bf bo = load_b(WcatT, KCAT, 2 * BH + c0, k0, lane);
    ao = __builtin_amdgcn_wmma_f32_16x16x32_bf16(false, a, false, bo, (short)0, ao, false, false);
    v16bf bg = load_b(WcatT, KCAT, 3 * BH + c0, k0, lane);
    ag = __builtin_amdgcn_wmma_f32_16x16x32_bf16(false, a, false, bg, (short)0, ag, false, false);
  }
  int nn = lane & 15, hi = lane >> 4;
  int col = c0 + nn;
  #pragma unroll
  for (int r = 0; r < 8; ++r) {
    int row = row0 + r + hi * 8;
    size_t xoff = ((size_t)row * BT + t) * FH;
    float pi = ai[r] + Xp[xoff + 0 * BH + col];
    float pf = af[r] + Xp[xoff + 1 * BH + col];
    float po = ao[r] + Xp[xoff + 2 * BH + col];
    float pg = ag[r] + Xp[xoff + 3 * BH + col];
    float gi = 1.f / (1.f + __expf(-pi));
    float gf = 1.f / (1.f + __expf(-pf));
    float go = 1.f / (1.f + __expf(-po));
    float gg = tanhf(pg);
    size_t hidx = (size_t)row * BH + col;
    float cn = gf * c[hidx] + gi * gg;
    float hn = go * tanhf(cn);
    c[hidx] = cn;
    h[hidx] = hn;
    hcat_out[(size_t)row * KCAT + col] = f2bf(hn);
    out[((size_t)row * BT + t) * BH + col] = hn;
  }
}

extern "C" void kernel_launch(void* const* d_in, const int* in_sizes, int n_in,
                              void* d_out, int out_size, void* d_ws, size_t ws_size,
                              hipStream_t stream) {
  (void)in_sizes; (void)n_in; (void)out_size; (void)ws_size;
  const float* x     = (const float*)d_in[0];   // (N,T,D)
  const float* A     = (const float*)d_in[1];   // (N,H,4,4) == (N,H,16)
  const float* Wx    = (const float*)d_in[2];   // (D,4H)
  const float* Wh    = (const float*)d_in[3];   // (H,4H)
  const float* Wattn = (const float*)d_in[4];   // (H,4H)
  const float* b     = (const float*)d_in[5];   // (4H,)
  float* out = (float*)d_out;                   // (N,T,H)

  char* ws = (char*)d_ws;
  float* Xp = (float*)ws;                        ws += (size_t)BN * BT * FH * 4;   // 512 MiB
  unsigned short* xbf   = (unsigned short*)ws;   ws += (size_t)BN * BT * BD * 2;   // 64 MiB
  unsigned short* WxT   = (unsigned short*)ws;   ws += (size_t)FH * BD * 2;        // 8 MiB
  unsigned short* WcatT = (unsigned short*)ws;   ws += (size_t)FH * KCAT * 2;      // 16 MiB
  unsigned short* hcat0 = (unsigned short*)ws;   ws += (size_t)BN * KCAT * 2;      // 1 MiB
  unsigned short* hcat1 = (unsigned short*)ws;   ws += (size_t)BN * KCAT * 2;      // 1 MiB
  float* hbuf = (float*)ws;                      ws += (size_t)BN * BH * 4;        // 1 MiB
  float* cbuf = (float*)ws;                      ws += (size_t)BN * BH * 4;        // 1 MiB

  // one-time weight transpose/convert (W^T layout -> contiguous K for B fragments)
  k_transpose_cvt<<<(FH * BD) / 256, 256, 0, stream>>>(Wx, WxT, BD, BD, 0);
  k_transpose_cvt<<<(FH * BH) / 256, 256, 0, stream>>>(Wh, WcatT, BH, KCAT, 0);
  k_transpose_cvt<<<(FH * BH) / 256, 256, 0, stream>>>(Wattn, WcatT, BH, KCAT, BH);
  // x -> bf16
  size_t xtot = (size_t)BN * BT * BD;
  k_cvt_bf16<<<(unsigned)((xtot + 255) / 256), 256, 0, stream>>>(x, xbf, xtot);
  // h0 = c0 = mean-pooled Af
  k_init<<<(BN * BH) / 256, 256, 0, stream>>>(A, hbuf, cbuf, hcat0);
  // big input projection (all timesteps at once)
  dim3 gproj(BN * BT / 256, FH / 64);
  k_xproj<<<gproj, 256, 0, stream>>>(xbf, WxT, b, Xp);
  // sequential recurrence
  dim3 ggate(BN / 128, BH / 16);
  for (int t = 0; t < BT; ++t) {
    unsigned short* hin  = (t & 1) ? hcat1 : hcat0;
    unsigned short* hout = (t & 1) ? hcat0 : hcat1;
    k_attn<<<BN, 256, 0, stream>>>(A, hbuf, hin);
    k_gate<<<ggate, 256, 0, stream>>>(hin, WcatT, Xp, hbuf, cbuf, hout, out, t);
  }
}
